// FlashGlm4moeLayer_47356309405778
// MI455X (gfx1250) — compile-verified, matrix operand order checked by reading, not run on hardware
//
#include <hip/hip_runtime.h>

// ---------------------------------------------------------------------------
// GLM4-MoE layer for MI455X (gfx1250, wave32, WMMA + TDM).
// All GEMMs run on v_wmma_f32_16x16x32_bf16 (fp32 accum). fp32 weights are
// converted to bf16 while staging into LDS; activations are kept bf16 in the
// workspace. A-tiles (plain strided bf16 copies) are staged with the Tensor
// Data Mover (tensor_load_to_lds + s_wait_tensorcnt); B-tiles need an
// f32->bf16 convert+transpose so they stay on the per-lane path. The main
// GEMM double-buffers LDS so the TDM DMA and weight conversion for tile k+1
// overlap the WMMAs of tile k. MoE computes only router-selected experts.
// ---------------------------------------------------------------------------

typedef __attribute__((ext_vector_type(16))) __bf16 bf16x16;
typedef __attribute__((ext_vector_type(8)))  float  f32x8;
typedef __attribute__((ext_vector_type(4)))  unsigned int u32x4;
typedef __attribute__((ext_vector_type(8)))  int  i32x8;
typedef __attribute__((ext_vector_type(4)))  int  i32x4;

#if __has_builtin(__builtin_amdgcn_tensor_load_to_lds)
#define USE_TDM 1
#else
#define USE_TDM 0
#endif

constexpr int B_   = 2;
constexpr int S_   = 1024;
constexpr int H_   = 2048;
constexpr int DH_  = 128;
constexpr int HQ_  = 16;
constexpr int HKV_ = 4;
constexpr int E_   = 16;
constexpr int K_   = 4;
constexpr int I_   = 1024;
constexpr int T_   = B_ * S_;

constexpr float EPS_        = 1e-5f;
constexpr float LOG_RBASE_  = 9.210340371976184f;    // ln(10000)
constexpr float ATTN_SCALE_ = 0.08838834764831845f;  // 1/sqrt(128)

// ---------------------------------------------------------------------------
// WMMA helpers
// ---------------------------------------------------------------------------
__device__ __forceinline__ f32x8 zero8() {
    f32x8 z;
#pragma unroll
    for (int i = 0; i < 8; ++i) z[i] = 0.0f;
    return z;
}

__device__ __forceinline__ f32x8 wmma_bf16(bf16x16 a, bf16x16 b, f32x8 c) {
    return __builtin_amdgcn_wmma_f32_16x16x32_bf16(
        false, a, false, b, (short)0, c, false, false);
}

// Load a 16x32 (bf16) A/B fragment from LDS per ISA 7.12.2:
// lane L: row = L&15, k-base = (L>>4)*8; elems 0..7 -> k0..k0+7,
// elems 8..15 -> k0+16..k0+23.  Two 16-byte LDS reads.
__device__ __forceinline__ bf16x16 lds_frag(const __bf16* base, int stride) {
    const int lane = threadIdx.x & 31;
    const __bf16* p = base + (lane & 15) * stride + ((lane >> 4) << 3);
    union { uint4 u[2]; bf16x16 v; } t;
    t.u[0] = *(const uint4*)(p);
    t.u[1] = *(const uint4*)(p + 16);
    return t.v;
}

// ---------------------------------------------------------------------------
// Tensor Data Mover: issue a 2D tile load of bf16 elements (rows x cols,
// row-major global with stride_elems between rows) into contiguous LDS.
// D# layout per CDNA5 ISA §8.3/8.4: group0 = {count, lds_addr, global_addr,
// type=2}; group1 = {data_size=2B, tensor_dim0/1, tile_dim0/1, dim0 stride}.
// One wave issues; EXEC is ignored for tensor ops. Tracked with TENSORcnt.
// ---------------------------------------------------------------------------
#if USE_TDM
__device__ __forceinline__ void tdm_load_tile_bf16(void* lds_ptr, const void* gptr,
                                                   int rows, int cols,
                                                   int stride_elems) {
    const unsigned lds_off = (unsigned)(size_t)lds_ptr;  // low 32 bits = LDS addr
    const unsigned long long ga = (unsigned long long)gptr;
    u32x4 g0;
    g0[0] = 1u;                                   // count=1, user descriptor
    g0[1] = lds_off;                              // lds_addr
    g0[2] = (unsigned)ga;                         // global_addr[31:0]
    g0[3] = (unsigned)((ga >> 32) & 0x01ffffffull) | (2u << 30);  // addr[56:32], type=2
    const unsigned td0 = (unsigned)stride_elems;  // tensor dim0 >= cols (no OOB)
    const unsigned td1 = 1u << 20;                // large: rows never OOB
    i32x8 g1;
    g1[0] = (int)(1u << 16);                      // data_size code 1 = 2 bytes
    g1[1] = (int)((td0 & 0xffffu) << 16);         // tensor_dim0[15:0]
    g1[2] = (int)(((td0 >> 16) & 0xffffu) | ((td1 & 0xffffu) << 16));
    g1[3] = (int)(((td1 >> 16) & 0xffffu) | (((unsigned)cols & 0xffffu) << 16));
    g1[4] = (int)((unsigned)rows & 0xffffu);      // tile_dim1 (tile_dim2 = 0)
    g1[5] = (int)td0;                             // tensor_dim0_stride[31:0]
    g1[6] = 0;
    g1[7] = 0;
    i32x4 g2 = {};
    i32x4 g3 = {};
#if defined(__clang_major__) && (__clang_major__ >= 23)
    i32x8 g4 = {};
    __builtin_amdgcn_tensor_load_to_lds(g0, g1, g2, g3, g4, 0);
#else
    __builtin_amdgcn_tensor_load_to_lds(g0, g1, g2, g3, 0);
#endif
}
#endif

// Manual fallback: stage a 64x32 bf16 A-tile from a bf16 global matrix.
__device__ __forceinline__ void load_a_tile(__bf16 (&As)[64][32],
                                            const __bf16* __restrict__ A,
                                            int lda, int mb, int kb, int tid) {
    const int row = tid >> 1, c0 = (tid & 1) * 16;
    const __bf16* src = A + (size_t)(mb + row) * lda + kb + c0;
    *(uint4*)&As[row][c0]     = *(const uint4*)(src);
    *(uint4*)&As[row][c0 + 8] = *(const uint4*)(src + 8);
}

// Stage (issue) a 64x32 bf16 A-tile: TDM if available, else per-lane copy.
__device__ __forceinline__ void stage_a_tile(__bf16 (&dst)[64][32],
                                             const __bf16* __restrict__ A,
                                             int lda, int mb, int kb,
                                             int tid, int wv) {
#if USE_TDM
    if (wv == 0)
        tdm_load_tile_bf16(&dst[0][0], A + (size_t)mb * lda + kb, 64, 32, lda);
    (void)tid;
#else
    (void)wv;
    load_a_tile(dst, A, lda, mb, kb, tid);
#endif
}

__device__ __forceinline__ void stage_a_wait(int wv) {
#if USE_TDM
    if (wv == 0) __builtin_amdgcn_s_wait_tensorcnt((short)0);
#else
    (void)wv;
#endif
}

// Stage a 64(n)x32(k) bf16 B-tile, transposed to [n][k], converting the fp32
// global weight matrix (row-major [k][n]) to bf16 on the fly.
__device__ __forceinline__ void load_b_tile(__bf16 (&Bs)[64][32],
                                            const float* __restrict__ Bm,
                                            int ldb, int kb, int nb, int tid) {
    const int k = tid >> 2, n0 = (tid & 3) * 16;
    const float* src = Bm + (size_t)(kb + k) * ldb + nb + n0;
#pragma unroll
    for (int i = 0; i < 16; ++i) Bs[n0 + i][k] = (__bf16)src[i];
}

// ---------------------------------------------------------------------------
// Generic GEMM: C[M,N](f32) = A(bf16)[M,K] @ B(f32->bf16)[K,N] (+bias)(+resid)
// Block = 128 threads (4 waves); 64x64 tile; wave w owns rows 16w..16w+15.
// Double-buffered: TDM (A) + convert (B) for tile k+1 overlap WMMAs of tile k.
// ---------------------------------------------------------------------------
template <bool BIAS, bool RESID>
__global__ __launch_bounds__(128) void gemm_bf16_kernel(
    const __bf16* __restrict__ A, int lda,
    const float* __restrict__ Bm, int ldb,
    const float* __restrict__ bias,
    const float* __restrict__ resid, int ldr,
    float* __restrict__ C, int ldc, int Kdim) {
    __shared__ __bf16 As[2][64][32];
    __shared__ __bf16 Bs[2][64][32];
    const int tid = threadIdx.x, wv = tid >> 5, lane = tid & 31;
    const int nb = blockIdx.x * 64, mb = blockIdx.y * 64;
    const int nk = Kdim / 32;

    f32x8 acc[4];
#pragma unroll
    for (int j = 0; j < 4; ++j) acc[j] = zero8();

    // prologue: stage tile 0
    stage_a_tile(As[0], A, lda, mb, 0, tid, wv);
    load_b_tile(Bs[0], Bm, ldb, 0, nb, tid);
    stage_a_wait(wv);
    __syncthreads();

    for (int kt = 0; kt < nk; ++kt) {
        const int cur = kt & 1, nxt = cur ^ 1;
        const bool more = (kt + 1 < nk);
        if (more) {
            stage_a_tile(As[nxt], A, lda, mb, (kt + 1) * 32, tid, wv);
            load_b_tile(Bs[nxt], Bm, ldb, (kt + 1) * 32, nb, tid);
            if (kt + 2 < nk)  // global_prefetch_b8 for the tile after next
                __builtin_prefetch(Bm + (size_t)((kt + 2) * 32 + (tid >> 2)) * ldb + nb, 0, 1);
        }
        bf16x16 af = lds_frag(&As[cur][16 * wv][0], 32);
#pragma unroll
        for (int j = 0; j < 4; ++j) {
            bf16x16 bf = lds_frag(&Bs[cur][16 * j][0], 32);
            acc[j] = wmma_bf16(af, bf, acc[j]);
        }
        if (more) {
            stage_a_wait(wv);
            __syncthreads();
        }
    }

#pragma unroll
    for (int j = 0; j < 4; ++j)
#pragma unroll
        for (int r = 0; r < 8; ++r) {
            const int m = mb + 16 * wv + r + ((lane & 16) ? 8 : 0);
            const int n = nb + 16 * j + (lane & 15);
            float v = acc[j][r];
            if (BIAS)  v += bias[n];
            if (RESID) v += resid[(size_t)m * ldr + n];
            C[(size_t)m * ldc + n] = v;
        }
}

// ---------------------------------------------------------------------------
// RMSNorm (f32 in) -> bf16 out, one block per token.
// ---------------------------------------------------------------------------
__global__ __launch_bounds__(256) void rmsnorm_bf16_kernel(
    const float* __restrict__ x, const float* __restrict__ w,
    __bf16* __restrict__ o) {
    const int t = blockIdx.x, tid = threadIdx.x;
    __shared__ float red[8];
    __shared__ float rs;
    float ss = 0.0f;
    for (int i = tid; i < H_; i += 256) {
        float v = x[(size_t)t * H_ + i];
        ss += v * v;
    }
#pragma unroll
    for (int m = 16; m; m >>= 1) ss += __shfl_xor(ss, m, 32);
    if ((tid & 31) == 0) red[tid >> 5] = ss;
    __syncthreads();
    if (tid == 0) {
        float s = 0.0f;
        for (int i = 0; i < 8; ++i) s += red[i];
        rs = rsqrtf(s / H_ + EPS_);
    }
    __syncthreads();
    for (int i = tid; i < H_; i += 256)
        o[(size_t)t * H_ + i] = (__bf16)(x[(size_t)t * H_ + i] * rs * w[i]);
}

// ---------------------------------------------------------------------------
// Per-head QK-RMSNorm + RoPE + relayout [T][NH][DH] -> [B][NH][S][DH] bf16.
// ROPE=false path is the plain V relayout/convert.
// ---------------------------------------------------------------------------
template <bool ROPE>
__global__ __launch_bounds__(128) void head_relayout_kernel(
    const float* __restrict__ in, const float* __restrict__ nw,
    __bf16* __restrict__ o, int NH) {
    const int hh = blockIdx.x, s = blockIdx.y, b = blockIdx.z, d = threadIdx.x;
    const int t = b * S_ + s;
    const float v = in[((size_t)t * NH + hh) * DH_ + d];
    float ov;
    if (ROPE) {
        __shared__ float row[DH_];
        __shared__ float red[4];
        __shared__ float rs;
        float ss = v * v;
#pragma unroll
        for (int m = 16; m; m >>= 1) ss += __shfl_xor(ss, m, 32);
        if ((d & 31) == 0) red[d >> 5] = ss;
        __syncthreads();
        if (d == 0) {
            float sm = red[0] + red[1] + red[2] + red[3];
            rs = rsqrtf(sm / DH_ + EPS_);
        }
        __syncthreads();
        const float xn = v * rs * nw[d];
        row[d] = xn;
        __syncthreads();
        const int   i  = d & 63;
        const float fr = (float)s * __expf(-(2.0f * i / (float)DH_) * LOG_RBASE_);
        const float c = cosf(fr), sn = sinf(fr);
        ov = (d < 64) ? (xn * c - row[d + 64] * sn)
                      : (xn * c + row[d - 64] * sn);
    } else {
        ov = v;
    }
    o[(((size_t)b * NH + hh) * S_ + s) * DH_ + d] = (__bf16)ov;
}

// ---------------------------------------------------------------------------
// Flash attention: one block per (q-tile of 64, head, batch). 4 waves; each
// wave owns 16 q-rows. Q*K^T and P*V via WMMA; online softmax with half-wave
// shuffle reductions (C/D layout: VGPR r -> row r/r+8, lane -> column).
// Q and K tiles staged via TDM; V is transposed per-lane (TDM can't).
// ---------------------------------------------------------------------------
__global__ __launch_bounds__(128) void attn_kernel(
    const __bf16* __restrict__ Q, const __bf16* __restrict__ Kg,
    const __bf16* __restrict__ V, __bf16* __restrict__ O) {
    __shared__ __bf16 Qs[64][128];
    __shared__ __bf16 Ks[64][128];
    __shared__ __bf16 Vst[128][64];   // V transposed: [dh][key]
    __shared__ __bf16 Ps[4][16][64];  // per-wave probability tile
    const int tid = threadIdx.x, wv = tid >> 5, lane = tid & 31;
    const int qt = blockIdx.x, h = blockIdx.y, b = blockIdx.z;
    const int qb = qt * 64;
    const int hk = h / (HQ_ / HKV_);

    {   // stage Q tile
        const __bf16* qg = Q + (((size_t)b * HQ_ + h) * S_ + qb) * DH_;
#if USE_TDM
        if (wv == 0) tdm_load_tile_bf16(&Qs[0][0], qg, 64, 128, DH_);
#else
        const int row = tid >> 1, c0 = (tid & 1) * 64;
        const uint4* src = (const uint4*)(qg + (size_t)row * DH_ + c0);
        uint4* dst = (uint4*)&Qs[row][c0];
#pragma unroll
        for (int i = 0; i < 8; ++i) dst[i] = src[i];
#endif
    }
    stage_a_wait(wv);
    __syncthreads();
    bf16x16 aq[4];
#pragma unroll
    for (int ks = 0; ks < 4; ++ks) aq[ks] = lds_frag(&Qs[16 * wv][32 * ks], 128);

    float mi[8], li[8];
    f32x8 accO[8];
#pragma unroll
    for (int r = 0; r < 8; ++r) { mi[r] = -3e38f; li[r] = 0.0f; }
#pragma unroll
    for (int j = 0; j < 8; ++j) accO[j] = zero8();

    for (int kt = 0; kt <= qt; ++kt) {
        __syncthreads();
        {   // stage K tile (TDM) and V tile (transposed, per-lane)
            const __bf16* kg = Kg + (((size_t)b * HKV_ + hk) * S_ + kt * 64) * DH_;
#if USE_TDM
            if (wv == 0) tdm_load_tile_bf16(&Ks[0][0], kg, 64, 128, DH_);
#else
            const int krow = tid >> 1, kc0 = (tid & 1) * 64;
            const uint4* ksrc = (const uint4*)(kg + (size_t)krow * DH_ + kc0);
            uint4* kdst = (uint4*)&Ks[krow][kc0];
#pragma unroll
            for (int i = 0; i < 8; ++i) kdst[i] = ksrc[i];
#endif
            const int row = tid >> 1, c0 = (tid & 1) * 64;
            const __bf16* vg = V + (((size_t)b * HKV_ + hk) * S_ + kt * 64) * DH_;
#pragma unroll
            for (int i = 0; i < 8; ++i) {
                union { uint4 u; __bf16 hv[8]; } uu;
                uu.u = ((const uint4*)(vg + (size_t)row * DH_ + c0))[i];
#pragma unroll
                for (int c = 0; c < 8; ++c) Vst[c0 + i * 8 + c][row] = uu.hv[c];
            }
        }
        stage_a_wait(wv);
        __syncthreads();

        // ---- scores S = Q K^T ----
        f32x8 sc[4];
#pragma unroll
        for (int j = 0; j < 4; ++j) sc[j] = zero8();
#pragma unroll
        for (int ks = 0; ks < 4; ++ks)
#pragma unroll
            for (int j = 0; j < 4; ++j) {
                bf16x16 bk = lds_frag(&Ks[16 * j][32 * ks], 128);
                sc[j] = wmma_bf16(aq[ks], bk, sc[j]);
            }

        // ---- scale + causal mask + online softmax ----
        const bool diag = (kt == qt);
        float mloc[8];
#pragma unroll
        for (int r = 0; r < 8; ++r) mloc[r] = -3e38f;
#pragma unroll
        for (int j = 0; j < 4; ++j)
#pragma unroll
            for (int r = 0; r < 8; ++r) {
                float sv = sc[j][r] * ATTN_SCALE_;
                if (diag) {
                    const int mg = qb + 16 * wv + r + ((lane & 16) ? 8 : 0);
                    const int ng = kt * 64 + 16 * j + (lane & 15);
                    if (ng > mg) sv = -3e38f;
                }
                sc[j][r] = sv;
                mloc[r] = fmaxf(mloc[r], sv);
            }
#pragma unroll
        for (int r = 0; r < 8; ++r)
#pragma unroll
            for (int msk = 1; msk < 16; msk <<= 1)
                mloc[r] = fmaxf(mloc[r], __shfl_xor(mloc[r], msk, 32));

        float alpha[8], psum[8];
#pragma unroll
        for (int r = 0; r < 8; ++r) {
            const float mn = fmaxf(mi[r], mloc[r]);
            alpha[r] = __expf(mi[r] - mn);
            mi[r] = mn;
            psum[r] = 0.0f;
        }
#pragma unroll
        for (int j = 0; j < 4; ++j)
#pragma unroll
            for (int r = 0; r < 8; ++r) {
                const float p = __expf(sc[j][r] - mi[r]);
                sc[j][r] = p;
                psum[r] += p;
            }
#pragma unroll
        for (int r = 0; r < 8; ++r) {
#pragma unroll
            for (int msk = 1; msk < 16; msk <<= 1)
                psum[r] += __shfl_xor(psum[r], msk, 32);
            li[r] = li[r] * alpha[r] + psum[r];
        }
#pragma unroll
        for (int j = 0; j < 8; ++j)
#pragma unroll
            for (int r = 0; r < 8; ++r) accO[j][r] *= alpha[r];

        // ---- P -> LDS (bf16), then O += P V via WMMA ----
#pragma unroll
        for (int j = 0; j < 4; ++j)
#pragma unroll
            for (int r = 0; r < 8; ++r) {
                const int ml = r + ((lane & 16) ? 8 : 0);
                Ps[wv][ml][16 * j + (lane & 15)] = (__bf16)sc[j][r];
            }
        // same-wave LDS write->read: wait on DS counter explicitly
        asm volatile("s_wait_dscnt 0" ::: "memory");
#pragma unroll
        for (int ks2 = 0; ks2 < 2; ++ks2) {
            bf16x16 ap = lds_frag(&Ps[wv][0][32 * ks2], 64);
#pragma unroll
            for (int j2 = 0; j2 < 8; ++j2) {
                bf16x16 bv = lds_frag(&Vst[16 * j2][32 * ks2], 64);
                accO[j2] = wmma_bf16(ap, bv, accO[j2]);
            }
        }
    }

    // epilogue: normalize, write o as [b][s][h][dh] == [T][HQ*DH]
#pragma unroll
    for (int j2 = 0; j2 < 8; ++j2)
#pragma unroll
        for (int r = 0; r < 8; ++r) {
            const int ml = 16 * wv + r + ((lane & 16) ? 8 : 0);
            const int n = 16 * j2 + (lane & 15);
            const float ov = accO[j2][r] / li[r];
            O[(((size_t)b * S_ + qb + ml) * HQ_ + h) * DH_ + n] = (__bf16)ov;
        }
}

// ---------------------------------------------------------------------------
// Router: one block per token. Recomputes post-attn RMSNorm in fp32, computes
// sigmoid gates, exact top-4 (bias added for selection only), normalized
// combine weights into dense cw[T][E].
// ---------------------------------------------------------------------------
__global__ __launch_bounds__(256) void router_kernel(
    const float* __restrict__ xa, const float* __restrict__ pw,
    const float* __restrict__ gw, const float* __restrict__ gb,
    float* __restrict__ cw) {
    const int t = blockIdx.x, tid = threadIdx.x;
    __shared__ float xr[H_];
    __shared__ float red[8];
    __shared__ float rs;
    __shared__ float logit[E_];
    float ss = 0.0f;
    for (int i = tid; i < H_; i += 256) {
        const float v = xa[(size_t)t * H_ + i];
        xr[i] = v;
        ss += v * v;
    }
#pragma unroll
    for (int m = 16; m; m >>= 1) ss += __shfl_xor(ss, m, 32);
    if ((tid & 31) == 0) red[tid >> 5] = ss;
    __syncthreads();
    if (tid == 0) {
        float s = 0.0f;
        for (int i = 0; i < 8; ++i) s += red[i];
        rs = rsqrtf(s / H_ + EPS_);
    }
    __syncthreads();
    {   // 16 threads per expert (one expert per wave half)
        const int e = tid >> 4, jj = tid & 15;
        float p = 0.0f;
        for (int i = jj; i < H_; i += 16)
            p += xr[i] * rs * pw[i] * gw[(size_t)e * H_ + i];
#pragma unroll
        for (int m = 1; m < 16; m <<= 1) p += __shfl_xor(p, m, 32);
        if (jj == 0) logit[e] = p;
    }
    __syncthreads();
    if (tid == 0) {
        float scr[E_], ch[E_];
        for (int e = 0; e < E_; ++e) {
            const float sg = 1.0f / (1.0f + __expf(-logit[e]));
            scr[e] = sg;
            ch[e] = sg + gb[e];
        }
        bool used[E_] = {};
        float outw[E_] = {};
        int sel[K_];
        float tsum = 0.0f;
        for (int k = 0; k < K_; ++k) {
            int bi = 0;
            float bv = -3e38f;
            for (int e = 0; e < E_; ++e)
                if (!used[e] && ch[e] > bv) { bv = ch[e]; bi = e; }
            used[bi] = true;
            sel[k] = bi;
            tsum += scr[bi];
        }
        const float inv = 1.0f / (tsum + 1e-20f);
        for (int k = 0; k < K_; ++k) outw[sel[k]] = scr[sel[k]] * inv;  // ROUTED_SCALE=1
        for (int e = 0; e < E_; ++e) cw[(size_t)t * E_ + e] = outw[e];
    }
}

// ---------------------------------------------------------------------------
// MoE stage A: inter[e][t][i] = silu(ht@wg_e) * (ht@wu_e) * cw[t][e] (bf16).
// e==16 is the shared expert (swg/swu, cw=1). Skips inactive (e, token-tile).
// ---------------------------------------------------------------------------
__global__ __launch_bounds__(128) void moe_up_kernel(
    const __bf16* __restrict__ A,
    const float* __restrict__ wg, const float* __restrict__ wu,
    const float* __restrict__ swg, const float* __restrict__ swu,
    const float* __restrict__ cw, __bf16* __restrict__ inter) {
    const int e = blockIdx.z;
    const int ib = blockIdx.x * 64, tb = blockIdx.y * 64;
    const int tid = threadIdx.x, wv = tid >> 5, lane = tid & 31;
    __shared__ float cws[64];
    __shared__ int act;
    if (tid == 0) act = (e == 16) ? 1 : 0;
    __syncthreads();
    if (tid < 64) {
        const float c = (e < 16) ? cw[(size_t)(tb + tid) * E_ + e] : 1.0f;
        cws[tid] = c;
        if (c != 0.0f) act = 1;
    }
    __syncthreads();
    if (!act) return;

    const float* Bg = (e < 16) ? wg + (size_t)e * H_ * I_ : swg;
    const float* Bu = (e < 16) ? wu + (size_t)e * H_ * I_ : swu;

    __shared__ __bf16 As[64][32];
    __shared__ __bf16 Bgs[64][32];
    __shared__ __bf16 Bus[64][32];
    f32x8 ag[4], au[4];
#pragma unroll
    for (int j = 0; j < 4; ++j) { ag[j] = zero8(); au[j] = zero8(); }

    for (int kb = 0; kb < H_; kb += 32) {
        __syncthreads();
        stage_a_tile(As, A, H_, tb, kb, tid, wv);
        load_b_tile(Bgs, Bg, I_, kb, ib, tid);
        load_b_tile(Bus, Bu, I_, kb, ib, tid);
        stage_a_wait(wv);
        __syncthreads();
        bf16x16 af = lds_frag(&As[16 * wv][0], 32);
#pragma unroll
        for (int j = 0; j < 4; ++j) {
            bf16x16 bg = lds_frag(&Bgs[16 * j][0], 32);
            ag[j] = wmma_bf16(af, bg, ag[j]);
            bf16x16 bu = lds_frag(&Bus[16 * j][0], 32);
            au[j] = wmma_bf16(af, bu, au[j]);
        }
    }

#pragma unroll
    for (int j = 0; j < 4; ++j)
#pragma unroll
        for (int r = 0; r < 8; ++r) {
            const int m = 16 * wv + r + ((lane & 16) ? 8 : 0);
            const int n = 16 * j + (lane & 15);
            const float g = ag[j][r], u = au[j][r];
            const float val = (g / (1.0f + __expf(-g))) * u * cws[m];
            inter[((size_t)e * T_ + tb + m) * I_ + ib + n] = (__bf16)val;
        }
}

// ---------------------------------------------------------------------------
// MoE stage B: out = x_attn + sum_e inter_e @ wd_e  (+ shared via swd).
// Accumulates all 17 experts in the same WMMA accumulators; skips inactive.
// ---------------------------------------------------------------------------
__global__ __launch_bounds__(128) void moe_down_kernel(
    const __bf16* __restrict__ inter,
    const float* __restrict__ wd, const float* __restrict__ swd,
    const float* __restrict__ cw, const float* __restrict__ resid,
    float* __restrict__ out) {
    const int nb = blockIdx.x * 64, tb = blockIdx.y * 64;
    const int tid = threadIdx.x, wv = tid >> 5, lane = tid & 31;
    __shared__ __bf16 As[64][32];
    __shared__ __bf16 Bds[64][32];
    __shared__ int act;

    f32x8 acc[4];
#pragma unroll
    for (int j = 0; j < 4; ++j) acc[j] = zero8();

    for (int e = 0; e < 17; ++e) {
        __syncthreads();
        if (tid == 0) act = (e == 16) ? 1 : 0;
        __syncthreads();
        if (e < 16 && tid < 64 && cw[(size_t)(tb + tid) * E_ + e] != 0.0f) act = 1;
        __syncthreads();
        if (!act) continue;

        const __bf16* Ae = inter + (size_t)e * T_ * I_;
        const float* Be = (e < 16) ? wd + (size_t)e * I_ * H_ : swd;
        for (int kb = 0; kb < I_; kb += 32) {
            __syncthreads();
            stage_a_tile(As, Ae, I_, tb, kb, tid, wv);
            load_b_tile(Bds, Be, H_, kb, nb, tid);
            stage_a_wait(wv);
            __syncthreads();
            bf16x16 af = lds_frag(&As[16 * wv][0], 32);
#pragma unroll
            for (int j = 0; j < 4; ++j) {
                bf16x16 bd = lds_frag(&Bds[16 * j][0], 32);
                acc[j] = wmma_bf16(af, bd, acc[j]);
            }
        }
    }

#pragma unroll
    for (int j = 0; j < 4; ++j)
#pragma unroll
        for (int r = 0; r < 8; ++r) {
            const int m = tb + 16 * wv + r + ((lane & 16) ? 8 : 0);
            const int n = nb + 16 * j + (lane & 15);
            out[(size_t)m * H_ + n] = resid[(size_t)m * H_ + n] + acc[j][r];
        }
}

// ---------------------------------------------------------------------------
// Host orchestration
// ---------------------------------------------------------------------------
extern "C" void kernel_launch(void* const* d_in, const int* in_sizes, int n_in,
                              void* d_out, int out_size, void* d_ws, size_t ws_size,
                              hipStream_t stream) {
    const float* x       = (const float*)d_in[0];
    const float* in_ln   = (const float*)d_in[1];
    const float* wq      = (const float*)d_in[2];
    const float* bq      = (const float*)d_in[3];
    const float* wk      = (const float*)d_in[4];
    const float* bk      = (const float*)d_in[5];
    const float* wv      = (const float*)d_in[6];
    const float* bv      = (const float*)d_in[7];
    const float* qn_w    = (const float*)d_in[8];
    const float* kn_w    = (const float*)d_in[9];
    const float* wo      = (const float*)d_in[10];
    const float* post_ln = (const float*)d_in[11];
    const float* gate_w  = (const float*)d_in[12];
    const float* gate_b  = (const float*)d_in[13];
    const float* wg      = (const float*)d_in[14];
    const float* wu      = (const float*)d_in[15];
    const float* wd      = (const float*)d_in[16];
    const float* swg     = (const float*)d_in[17];
    const float* swu     = (const float*)d_in[18];
    const float* swd     = (const float*)d_in[19];
    float* out = (float*)d_out;

    char* ws = (char*)d_ws;
    size_t off = 0;
    auto wsalloc = [&](size_t bytes) -> void* {
        void* p = ws + off;
        off += (bytes + 255) & ~(size_t)255;
        return p;
    };
    __bf16* h1    = (__bf16*)wsalloc((size_t)T_ * H_ * 2);
    float*  qf    = (float*)wsalloc((size_t)T_ * HQ_ * DH_ * 4);
    float*  kf    = (float*)wsalloc((size_t)T_ * HKV_ * DH_ * 4);
    float*  vf    = (float*)wsalloc((size_t)T_ * HKV_ * DH_ * 4);
    __bf16* q_bf  = (__bf16*)wsalloc((size_t)T_ * HQ_ * DH_ * 2);
    __bf16* k_bf  = (__bf16*)wsalloc((size_t)T_ * HKV_ * DH_ * 2);
    __bf16* v_bf  = (__bf16*)wsalloc((size_t)T_ * HKV_ * DH_ * 2);
    __bf16* o_bf  = (__bf16*)wsalloc((size_t)T_ * HQ_ * DH_ * 2);
    float*  xa    = (float*)wsalloc((size_t)T_ * H_ * 4);
    __bf16* h2    = (__bf16*)wsalloc((size_t)T_ * H_ * 2);
    float*  cwp   = (float*)wsalloc((size_t)T_ * E_ * 4);
    __bf16* inter = (__bf16*)wsalloc((size_t)17 * T_ * I_ * 2);
    (void)ws_size; (void)in_sizes; (void)n_in; (void)out_size;

    // 1) input RMSNorm -> bf16
    rmsnorm_bf16_kernel<<<T_, 256, 0, stream>>>(x, in_ln, h1);

    // 2) QKV projections (bias), f32 intermediates
    gemm_bf16_kernel<true, false><<<dim3((HQ_ * DH_) / 64, T_ / 64), 128, 0, stream>>>(
        h1, H_, wq, HQ_ * DH_, bq, nullptr, 0, qf, HQ_ * DH_, H_);
    gemm_bf16_kernel<true, false><<<dim3((HKV_ * DH_) / 64, T_ / 64), 128, 0, stream>>>(
        h1, H_, wk, HKV_ * DH_, bk, nullptr, 0, kf, HKV_ * DH_, H_);
    gemm_bf16_kernel<true, false><<<dim3((HKV_ * DH_) / 64, T_ / 64), 128, 0, stream>>>(
        h1, H_, wv, HKV_ * DH_, bv, nullptr, 0, vf, HKV_ * DH_, H_);

    // 3) QK norm + RoPE + relayout; V relayout
    head_relayout_kernel<true><<<dim3(HQ_, S_, B_), 128, 0, stream>>>(qf, qn_w, q_bf, HQ_);
    head_relayout_kernel<true><<<dim3(HKV_, S_, B_), 128, 0, stream>>>(kf, kn_w, k_bf, HKV_);
    head_relayout_kernel<false><<<dim3(HKV_, S_, B_), 128, 0, stream>>>(vf, nullptr, v_bf, HKV_);

    // 4) flash attention
    attn_kernel<<<dim3(S_ / 64, HQ_, B_), 128, 0, stream>>>(q_bf, k_bf, v_bf, o_bf);

    // 5) output projection + residual
    gemm_bf16_kernel<false, true><<<dim3(H_ / 64, T_ / 64), 128, 0, stream>>>(
        o_bf, HQ_ * DH_, wo, H_, nullptr, x, H_, xa, H_, HQ_ * DH_);

    // 6) post-attn RMSNorm -> bf16; router (fp32, recomputes norm internally)
    rmsnorm_bf16_kernel<<<T_, 256, 0, stream>>>(xa, post_ln, h2);
    router_kernel<<<T_, 256, 0, stream>>>(xa, post_ln, gate_w, gate_b, cwp);

    // 7) MoE up (16 routed + 1 shared), then down + residual -> out
    moe_up_kernel<<<dim3(I_ / 64, T_ / 64, 17), 128, 0, stream>>>(
        h2, wg, wu, swg, swu, cwp, inter);
    moe_down_kernel<<<dim3(H_ / 64, T_ / 64), 128, 0, stream>>>(
        inter, wd, swd, cwp, xa, out);
}
